// RVLinearlayer_3272765079980
// MI455X (gfx1250) — matrix operand order repaired and checked by reading, not compiled, not run on hardware
//
#include <hip/hip_runtime.h>
#include <math.h>

// ---------------------------------------------------------------------------
// RVLinearlayer for MI455X (gfx1250, wave32, WMMA).
//   mu_out    = mu_in @ w_mu + b_mu
//   sigma_out = [sigma_in | mu_in^2] @ [[softplus(w_sigma)^T + w_mu^2];[softplus(w_sigma)^T]] + Bsig
//   + two scalar KL reductions.
// bf16 WMMA GEMMs; global->LDS staging via CDNA5 async copies (ASYNCcnt),
// which keeps the per-lane VGPR count ~120 (no scratch spills).
// ---------------------------------------------------------------------------

typedef __attribute__((ext_vector_type(16))) __bf16 v16bf;
typedef __attribute__((ext_vector_type(8)))  __bf16 v8bf;
typedef __attribute__((ext_vector_type(8)))  float  v8f;

#define TILE_M 128
#define TILE_N 128
#define TILE_K 32
#define LDT    40   // padded LDS k-stride (elements); 80B row stride, 16B aligned, bank-conflict free

__device__ __forceinline__ unsigned short f32_to_bf16(float f) {
  unsigned u = __float_as_uint(f);
  u += 0x7FFFu + ((u >> 16) & 1u);   // round-to-nearest-even
  return (unsigned short)(u >> 16);
}

__device__ __forceinline__ float block_reduce_sum(float v, float* sbuf) {
  const int t = threadIdx.x;
  sbuf[t] = v;
  __syncthreads();
  for (int s = 128; s > 0; s >>= 1) {
    if (t < s) sbuf[t] += sbuf[t + s];
    __syncthreads();
  }
  float r = sbuf[0];
  __syncthreads();
  return r;
}

// low 32 bits of an aperture-based generic pointer to __shared__ = LDS offset
__device__ __forceinline__ unsigned lds_addr_of(const void* p) {
  return (unsigned)(unsigned long long)p;
}

// ---------------------------------------------------------------------------
// Prep: activations -> bf16.  A_mu[b][k] = mu_in; A_s[b][0:IN]=sigma_in, A_s[b][IN:2IN]=mu_in^2
// ---------------------------------------------------------------------------
__global__ __launch_bounds__(256) void prep_inputs(
    const float* __restrict__ mu_in, const float* __restrict__ sigma_in,
    unsigned short* __restrict__ A_mu, unsigned short* __restrict__ A_s, int IN) {
  const int i = blockIdx.x * 256 + threadIdx.x;   // exact grid, no bounds check
  const int b = i / IN;
  const int k = i - b * IN;
  const float m = mu_in[i];
  const float s = sigma_in[i];
  A_mu[i] = f32_to_bf16(m);
  const size_t base = (size_t)b * (size_t)(2 * IN);
  A_s[base + k]      = f32_to_bf16(s);
  A_s[base + IN + k] = f32_to_bf16(m * m);
}

// ---------------------------------------------------------------------------
// Prep: weights -> bf16, N-major (row n, contiguous k) so GEMM B-tiles coalesce.
//   B_mu[o][i]    = w_mu[i][o]
//   B_s [o][i]    = softplus(w_sigma[o][i]) + 1e-6 + w_mu[i][o]^2
//   B_s [o][IN+i] = softplus(w_sigma[o][i]) + 1e-6
// Also block-reduces w_kl partials: S1 = sum(w_mu^2), S2 = sum(1 - Ws + log Ws).
// ---------------------------------------------------------------------------
__global__ __launch_bounds__(256) void prep_weights(
    const float* __restrict__ w_mu, const float* __restrict__ w_sigma,
    unsigned short* __restrict__ B_mu, unsigned short* __restrict__ B_s,
    float* __restrict__ partials, int IN, int OUT, int nBlocks) {
  __shared__ float sbuf[256];
  const int idx = blockIdx.x * 256 + threadIdx.x;   // = o*IN + i, exact grid
  const int o = idx / IN;
  const int i = idx - o * IN;
  const float wm = w_mu[(size_t)i * OUT + o];
  const float ws = log1pf(expf(w_sigma[idx])) + 1e-6f;
  B_mu[idx] = f32_to_bf16(wm);
  const size_t b2 = (size_t)o * (size_t)(2 * IN);
  B_s[b2 + i]      = f32_to_bf16(ws + wm * wm);
  B_s[b2 + IN + i] = f32_to_bf16(ws);
  const float s1 = block_reduce_sum(wm * wm, sbuf);
  const float s2 = block_reduce_sum(1.0f - ws + logf(ws), sbuf);
  if (threadIdx.x == 0) {
    partials[blockIdx.x]           = s1;
    partials[nBlocks + blockIdx.x] = s2;
  }
}

// ---------------------------------------------------------------------------
// Finalize KL scalars + sigma-bias vector (single block).
//   w_kl = 0.5*(OUT*S1 - IN*S2)        (norm_sq broadcast over OUT rows; k = IN)
//   b_kl = 0.5*(sum(b_mu^2) - OUT*sum(1 - Bs + log Bs))   (k = OUT)
// ---------------------------------------------------------------------------
__global__ __launch_bounds__(256) void finalize_kl(
    const float* __restrict__ partials, int nPart,
    const float* __restrict__ b_mu, const float* __restrict__ b_sigma,
    float* __restrict__ bias_sig, float* __restrict__ kl_out, int IN, int OUT) {
  __shared__ float sbuf[256];
  float s1 = 0.f, s2 = 0.f, t1 = 0.f, t2 = 0.f;
  for (int i = threadIdx.x; i < nPart; i += 256) {
    s1 += partials[i];
    s2 += partials[nPart + i];
  }
  for (int n = threadIdx.x; n < OUT; n += 256) {
    const float bs = log1pf(expf(b_sigma[n])) + 1e-6f;
    bias_sig[n] = bs;
    const float bm = b_mu[n];
    t1 += bm * bm;
    t2 += 1.0f - bs + logf(bs);
  }
  s1 = block_reduce_sum(s1, sbuf);
  s2 = block_reduce_sum(s2, sbuf);
  t1 = block_reduce_sum(t1, sbuf);
  t2 = block_reduce_sum(t2, sbuf);
  if (threadIdx.x == 0) {
    kl_out[0] = 0.5f * ((float)OUT * s1 - (float)IN * s2);
    kl_out[1] = 0.5f * (t1 - (float)OUT * t2);
  }
}

// ---------------------------------------------------------------------------
// bf16 WMMA GEMM:  C[M][N] = A[M][K] * B^T  (B stored N-major: Bm[n][k]) + bias[n]
// 256 threads = 8 waves; block tile 128x128; per-wave 32x64 (2x4 WMMA tiles);
// K-step 32; double-buffered LDS fed by global_load_async_to_lds_b128 (ASYNCcnt).
// ---------------------------------------------------------------------------
__global__ __launch_bounds__(256) void gemm_bf16_wmma(
    const unsigned short* __restrict__ A,
    const unsigned short* __restrict__ Bm,
    const float* __restrict__ bias,
    float* __restrict__ C,
    int M, int N, int K) {
  __shared__ unsigned short As[2][TILE_M * LDT];
  __shared__ unsigned short Bs[2][TILE_N * LDT];

  const int tid      = threadIdx.x;
  const int lane     = tid & 31;
  const int laneHalf = lane >> 4;   // 0|1
  const int lane16   = lane & 15;
  const int wave     = tid >> 5;

  const int blockM = blockIdx.y * TILE_M;
  const int blockN = blockIdx.x * TILE_N;
  const int waveM  = (wave >> 1) * 32;   // 4 waves along M
  const int waveN  = (wave & 1) * 64;    // 2 waves along N

  v8f acc[2][4] = {};

  // staging geometry: each thread moves 2x16B per matrix per stage (async, no VGPR data)
  const int ldRow = tid >> 2;        // 0..63
  const int ldCol = (tid & 3) * 8;   // 0,8,16,24
  const unsigned short* gA = A  + (size_t)blockM * K;
  const unsigned short* gB = Bm + (size_t)blockN * K;

  // Issue one stage of global->LDS async copies: 4 x b128 per thread (2 A + 2 B).
  auto issue_stage = [&](int kt, int buf) {
#pragma unroll
    for (int p = 0; p < 2; ++p) {
      const int row = ldRow + p * 64;
      const unsigned voff = (unsigned)(((unsigned)row * (unsigned)K + (unsigned)kt +
                                        (unsigned)ldCol) * 2u);  // byte offset, same for A and B
      const unsigned ldsA = lds_addr_of(&As[buf][row * LDT + ldCol]);
      const unsigned ldsB = lds_addr_of(&Bs[buf][row * LDT + ldCol]);
      asm volatile("global_load_async_to_lds_b128 %0, %1, %2"
                   :: "v"(ldsA), "v"(voff), "s"(gA) : "memory");
      asm volatile("global_load_async_to_lds_b128 %0, %1, %2"
                   :: "v"(ldsB), "v"(voff), "s"(gB) : "memory");
    }
  };

  const int nStages = K / TILE_K;
  issue_stage(0, 0);

  for (int s = 0; s < nStages; ++s) {
    const int cur = s & 1;
    if (s + 1 < nStages) {
      issue_stage((s + 1) * TILE_K, cur ^ 1);        // prefetch next stage
      asm volatile("s_wait_asynccnt 0x4" ::: "memory");  // stage s done, next 4 in flight
    } else {
      asm volatile("s_wait_asynccnt 0x0" ::: "memory");
    }
    __syncthreads();   // stage-s tile visible to all waves

    // A fragments (16x32): lane half selects K {0..7,16..23} vs {8..15,24..31}
    v16bf afrag[2];
#pragma unroll
    for (int mi = 0; mi < 2; ++mi) {
      const unsigned short* p = &As[cur][(waveM + mi * 16 + lane16) * LDT + laneHalf * 8];
      v8bf lo = *(const v8bf*)(p);
      v8bf hi = *(const v8bf*)(p + 16);
      afrag[mi] = __builtin_shufflevector(lo, hi, 0, 1, 2, 3, 4, 5, 6, 7,
                                          8, 9, 10, 11, 12, 13, 14, 15);
    }
    // B fragments (32x16): lane half selects K {0..7,16..23} vs {8..15,24..31}
    v16bf bfrag[4];
#pragma unroll
    for (int ni = 0; ni < 4; ++ni) {
      const unsigned short* p = &Bs[cur][(waveN + ni * 16 + lane16) * LDT + laneHalf * 8];
      v8bf lo = *(const v8bf*)(p);
      v8bf hi = *(const v8bf*)(p + 16);
      bfrag[ni] = __builtin_shufflevector(lo, hi, 0, 1, 2, 3, 4, 5, 6, 7,
                                          8, 9, 10, 11, 12, 13, 14, 15);
    }

#pragma unroll
    for (int mi = 0; mi < 2; ++mi)
#pragma unroll
      for (int ni = 0; ni < 4; ++ni)
        acc[mi][ni] = __builtin_amdgcn_wmma_f32_16x16x32_bf16(
            false, afrag[mi], false, bfrag[ni], (short)0, acc[mi][ni], false, false);

    __syncthreads();   // all waves done reading buf `cur` before it is re-filled
  }

  // Epilogue: C/D layout — VGPR r holds M = r + 8*laneHalf, N = lane16.
#pragma unroll
  for (int mi = 0; mi < 2; ++mi) {
    const int rowBase = blockM + waveM + mi * 16 + laneHalf * 8;
#pragma unroll
    for (int ni = 0; ni < 4; ++ni) {
      const int col = blockN + waveN + ni * 16 + lane16;
      const float bv = bias[col];
#pragma unroll
      for (int r = 0; r < 8; ++r) {
        C[(size_t)(rowBase + r) * N + col] = acc[mi][ni][r] + bv;
      }
    }
  }
}

// ---------------------------------------------------------------------------
extern "C" void kernel_launch(void* const* d_in, const int* in_sizes, int n_in,
                              void* d_out, int out_size, void* d_ws, size_t ws_size,
                              hipStream_t stream) {
  const float* mu_in    = (const float*)d_in[0];
  const float* sigma_in = (const float*)d_in[1];
  const float* w_mu     = (const float*)d_in[2];   // (IN, OUT)
  const float* w_sigma  = (const float*)d_in[3];   // (OUT, IN)
  const float* b_mu     = (const float*)d_in[4];   // (OUT)
  const float* b_sigma  = (const float*)d_in[5];   // (OUT)
  float* out = (float*)d_out;

  const int Bt = 4096, IN = 2048, OUT = 2048;

  // workspace layout (~76 MB)
  unsigned short* A_mu = (unsigned short*)d_ws;                 // Bt*IN bf16
  unsigned short* A_s  = A_mu + (size_t)Bt * IN;                // Bt*2IN bf16
  unsigned short* B_mu = A_s  + (size_t)Bt * 2 * IN;            // OUT*IN bf16 (N-major)
  unsigned short* B_s  = B_mu + (size_t)OUT * IN;               // OUT*2IN bf16 (N-major)
  float* bias_sig = (float*)(B_s + (size_t)OUT * 2 * IN);       // OUT f32
  float* partials = bias_sig + OUT;                             // 2 * 16384 f32

  float* mu_out    = out;
  float* sigma_out = out + (size_t)Bt * OUT;
  float* kl_out    = out + (size_t)2 * Bt * OUT;                // [w_kl, b_kl]

  prep_inputs<<<(Bt * IN) / 256, 256, 0, stream>>>(mu_in, sigma_in, A_mu, A_s, IN);

  const int wBlocks = (IN * OUT) / 256;  // 16384
  prep_weights<<<wBlocks, 256, 0, stream>>>(w_mu, w_sigma, B_mu, B_s, partials, IN, OUT, wBlocks);

  finalize_kl<<<1, 256, 0, stream>>>(partials, wBlocks, b_mu, b_sigma, bias_sig, kl_out, IN, OUT);

  dim3 grid(OUT / TILE_N, Bt / TILE_M);  // (16, 32)
  gemm_bf16_wmma<<<grid, 256, 0, stream>>>(A_mu, B_mu, b_mu, mu_out, Bt, OUT, IN);
  gemm_bf16_wmma<<<grid, 256, 0, stream>>>(A_s, B_s, bias_sig, sigma_out, Bt, OUT, 2 * IN);
}
